// ClientModel_66434554134739
// MI455X (gfx1250) — matrix operand order repaired and checked by reading, not compiled, not run on hardware
//
#include <hip/hip_runtime.h>
#include <hip/hip_bf16.h>
#include <stdint.h>

// ---------------------------------------------------------------------------
// Problem constants (match reference)
// ---------------------------------------------------------------------------
#define HID   512
#define BATCH 128
#define SEQT  64
#define VOCAB 10000
#define G4    2048          // 4*HID
#define KCAT  1024          // HID (x) + HID (h)
#define KSTEPS (KCAT / 32)  // 32 WMMA K-steps in the LSTM GEMM
#define NTILES (VOCAB / 16) // 625 vocab tiles
#define LPAD  520           // padded LDS row length (uint16): 1040B -> conflict-free b128 reads

typedef __attribute__((ext_vector_type(16))) __bf16 v16bf;
typedef __attribute__((ext_vector_type(8)))  float  v8f;

union AFrag { uint4 u[2]; v16bf v; };

static __device__ __forceinline__ uint16_t f2bf(float f) {
    union { float f; uint32_t u; } x; x.f = f;
    uint32_t r = x.u + 0x7FFFu + ((x.u >> 16) & 1u);   // round-to-nearest-even
    return (uint16_t)(r >> 16);
}
static __device__ __forceinline__ float sigmoidf_(float x) {
    return 1.0f / (1.0f + __expf(-x));
}

// ---------------------------------------------------------------------------
// Pack kernels: fp32 weights -> bf16, transposed to [N, K] row-major so the
// WMMA B-fragment loads are contiguous 16-byte chunks.
// ---------------------------------------------------------------------------
__global__ void pack_lstm_weights(const float* __restrict__ Wx,   // [HID, 2048]
                                  const float* __restrict__ Wh,   // [HID, 2048]
                                  uint16_t* __restrict__ WT)      // [2048, 1024]
{
    int idx = blockIdx.x * blockDim.x + threadIdx.x;              // 2048*1024
    if (idx >= G4 * KCAT) return;
    int k = idx & (KCAT - 1);
    int n = idx >> 10;
    float v = (k < HID) ? Wx[(size_t)k * G4 + n] : Wh[(size_t)(k - HID) * G4 + n];
    WT[idx] = f2bf(v);
}

__global__ void pack_softmax_w(const float* __restrict__ W,       // [HID, VOCAB]
                               uint16_t* __restrict__ SWT)        // [VOCAB, HID]
{
    int idx = blockIdx.x * blockDim.x + threadIdx.x;              // 10000*512
    if (idx >= VOCAB * HID) return;
    int k = idx & (HID - 1);
    int n = idx >> 9;
    SWT[idx] = f2bf(W[(size_t)k * VOCAB + n]);
}

__global__ void embed_gather(const int* __restrict__ feat,        // [B, T]
                             const float* __restrict__ emb,       // [VOCAB, HID]
                             uint16_t* __restrict__ xbf)          // [T, B, HID]
{
    int idx = blockIdx.x * blockDim.x + threadIdx.x;              // T*B*HID
    if (idx >= SEQT * BATCH * HID) return;
    int h  = idx & (HID - 1);
    int bt = idx >> 9;
    int b  = bt & (BATCH - 1);
    int t  = bt >> 7;
    xbf[idx] = f2bf(emb[(size_t)feat[b * SEQT + t] * HID + h]);
}

__global__ void zero_u32(uint32_t* __restrict__ p, int n) {
    int idx = blockIdx.x * blockDim.x + threadIdx.x;
    if (idx < n) p[idx] = 0u;
}

// ---------------------------------------------------------------------------
// Fused LSTM cell: GEMM + bias + gate nonlinearities + length masking.
//
// Each wave owns 16 batch rows x 16 hidden columns, computing those columns
// across ALL FOUR gate blocks (N = g*512 + j0 + col), so the gate epilogue
// runs in-register with no [128,2048] intermediate.
//
// The K-loop is a branch-free software pipeline: the loop over 32 K-steps is
// FULLY unrolled, so the "load step s+1, then WMMA step s" ordering becomes
// one straight-line block -- every guard and the x/h base select fold to
// compile-time constants (round 3's partial unroll left dynamic branches).
//
// Block: 256 threads = 8 waves (rows 0..127). Grid: 512/16 = 32 blocks.
// ---------------------------------------------------------------------------
__global__ __launch_bounds__(256)
void lstm_cell(const uint16_t* __restrict__ X,     // [128, 512] bf16 (K 0..511)
               const uint16_t* __restrict__ Hin,   // [128, 512] bf16 (K 512..1023)
               const uint16_t* __restrict__ WT,    // [2048, 1024] bf16 (N,K)
               const float* __restrict__ bias,     // [2048]
               float* __restrict__ h,              // [128, 512] fp32 state
               float* __restrict__ c,              // [128, 512] fp32 state
               uint16_t* __restrict__ Hout,        // [128, 512] bf16 masked h (next step)
               uint16_t* __restrict__ out_bf,      // [128, 512] bf16 m*h_new (next layer)
               const int* __restrict__ seqlen,     // [128]
               int t)
{
    const int wave  = threadIdx.x >> 5;
    const int lane  = threadIdx.x & 31;
    const int j0    = blockIdx.x * 16;        // hidden-column tile
    const int mrow0 = wave * 16;              // batch-row tile

    // A fragment (16x32 bf16): m = lane&15, K half-offset 0/16
    const int am = lane & 15;
    const int kb = (lane < 16) ? 0 : 16;
    // B fragment (32x16 bf16): n = lane&15, K half-offset 0/16
    const int bn  = lane & 15;
    const int bko = (lane < 16) ? 0 : 16;

    v8f acc[4];
#pragma unroll
    for (int g = 0; g < 4; ++g) { v8f z = {}; acc[g] = z; }

    const uint16_t* xrow = X   + (size_t)(mrow0 + am) * HID;
    const uint16_t* hrow = Hin + (size_t)(mrow0 + am) * HID;
    const uint16_t* brow[4];
#pragma unroll
    for (int g = 0; g < 4; ++g)
        brow[g] = WT + (size_t)(g * HID + j0 + bn) * KCAT + bko;

    // -------- branch-free software-pipelined K loop (fully unrolled) -------
    AFrag fa[2];
    AFrag fb[2][4];

    // Preload step 0 (k = 0, from X)
    fa[0].u[0] = *(const uint4*)(xrow + kb);
    fa[0].u[1] = *(const uint4*)(xrow + 8 + kb);
#pragma unroll
    for (int g = 0; g < 4; ++g) {
        fb[0][g].u[0] = *(const uint4*)(brow[g]);
        fb[0][g].u[1] = *(const uint4*)(brow[g] + 8);
    }

#pragma unroll
    for (int step = 0; step < KSTEPS; ++step) {
        const int cur = step & 1;
        const int nxt = cur ^ 1;

        if (step + 1 < KSTEPS) {              // compile-time after full unroll
            const int kn = (step + 1) * 32;
            const uint16_t* arow = (kn < HID) ? (xrow + kn) : (hrow + (kn - HID));
            fa[nxt].u[0] = *(const uint4*)(arow + kb);
            fa[nxt].u[1] = *(const uint4*)(arow + 8 + kb);
#pragma unroll
            for (int g = 0; g < 4; ++g) {
                fb[nxt][g].u[0] = *(const uint4*)(brow[g] + kn);
                fb[nxt][g].u[1] = *(const uint4*)(brow[g] + kn + 8);
            }
            if ((kn & 63) == 0 && kn + 64 < KCAT) {
                __builtin_prefetch(brow[0] + kn + 64, 0, 1);
                __builtin_prefetch(brow[1] + kn + 64, 0, 1);
                __builtin_prefetch(brow[2] + kn + 64, 0, 1);
                __builtin_prefetch(brow[3] + kn + 64, 0, 1);
            }
        }

#pragma unroll
        for (int g = 0; g < 4; ++g)
            acc[g] = __builtin_amdgcn_wmma_f32_16x16x32_bf16(
                false, fa[cur].v, false, fb[cur][g].v, (short)0, acc[g], false, false);
    }

    // Epilogue: acc[0]=i, acc[1]=cg, acc[2]=f, acc[3]=o for (16 rows x 16 cols).
    const int col  = lane & 15;
    const int mofs = (lane < 16) ? 0 : 8;
    const float bi = bias[j0 + col];
    const float bc = bias[HID + j0 + col];
    const float bg = bias[2 * HID + j0 + col];
    const float bo = bias[3 * HID + j0 + col];

#pragma unroll
    for (int vv = 0; vv < 8; ++vv) {
        const int brw = mrow0 + vv + mofs;               // batch index
        const float m = (t < seqlen[brw]) ? 1.0f : 0.0f;
        const int idx = brw * HID + j0 + col;

        float gi = acc[0][vv] + bi;
        float gc = acc[1][vv] + bc;
        float gf = acc[2][vv] + bg;
        float go = acc[3][vv] + bo;

        float cold = c[idx];
        float hold = h[idx];
        float cn = sigmoidf_(gf) * cold + sigmoidf_(gi) * tanhf(gc);
        float hn = sigmoidf_(go) * tanhf(cn);

        float hm = m * hn + (1.0f - m) * hold;
        float cm = m * cn + (1.0f - m) * cold;
        h[idx] = hm;
        c[idx] = cm;
        Hout[idx]   = f2bf(hm);
        out_bf[idx] = f2bf(m * hn);
    }
}

// ---------------------------------------------------------------------------
// Fused logits + log-softmax cross-entropy (FLOP-dominant kernel).
//
// 4 waves per block share each 16-column vocab tile of SWT through LDS
// (double-buffered: registers stage tile nt+1 from global while tile nt is
// consumed from LDS), cutting L2 traffic 4x and hiding global latency.
// K-loop fully unrolled: 16 straight-line WMMAs per tile so ds/global loads
// interleave freely with the WMMA stream. Online log-sum-exp per lane; final
// 16-lane shfl merge. Never materializes the [8192,10000] logits.
// ---------------------------------------------------------------------------
__global__ __launch_bounds__(128)
void logits_xent(const uint16_t* __restrict__ outs,     // [T*B, 512] bf16, r=t*128+b
                 const uint16_t* __restrict__ SWT,      // [VOCAB, 512] bf16
                 const float* __restrict__ sb,          // [VOCAB]
                 const int* __restrict__ labels,        // [B, T]
                 float* __restrict__ xent)              // [T*B]
{
    __shared__ uint16_t smem[2][16 * LPAD];             // 2 x 16.6 KB

    const int tid  = threadIdx.x;
    const int wave = tid >> 5;
    const int lane = tid & 31;
    const int r0   = (blockIdx.x * 4 + wave) * 16;      // row tile base

    const int am   = lane & 15;
    const int kb   = (lane < 16) ? 0 : 16;
    const int bn   = lane & 15;
    const int bko  = (lane < 16) ? 0 : 16;
    const int mofs = (lane < 16) ? 0 : 8;

    int mylab[8];
#pragma unroll
    for (int vv = 0; vv < 8; ++vv) {
        int r = r0 + vv + mofs;
        int b = r & (BATCH - 1);
        int t = r >> 7;
        mylab[vv] = labels[b * SEQT + t];
    }

    float rmax[8], rsum[8], rlab[8];
#pragma unroll
    for (int vv = 0; vv < 8; ++vv) { rmax[vv] = -1e30f; rsum[vv] = 0.0f; rlab[vv] = -1e30f; }

    const uint16_t* arow = outs + (size_t)(r0 + am) * HID;

    // Tile staging: 16 rows x 512 cols bf16 = 1024 uint4; 8 uint4 per thread.
    uint4 stage[8];
    {
        const uint16_t* src = SWT;                      // tile 0
#pragma unroll
        for (int i = 0; i < 8; ++i)
            stage[i] = *(const uint4*)(src + (size_t)(tid * 8 + i) * 8);
    }
#pragma unroll
    for (int i = 0; i < 8; ++i) {
        int q = tid * 8 + i;
        *(uint4*)&smem[0][(q >> 6) * LPAD + (q & 63) * 8] = stage[i];
    }
    __syncthreads();

    for (int nt = 0; nt < NTILES; ++nt) {
        const int cur = nt & 1;

        if (nt + 1 < NTILES) {                          // prefetch next tile to regs
            const uint16_t* src = SWT + (size_t)(nt + 1) * 16 * HID;
#pragma unroll
            for (int i = 0; i < 8; ++i)
                stage[i] = *(const uint4*)(src + (size_t)(tid * 8 + i) * 8);
        }

        v8f acc = {};
        const uint16_t* bbase = &smem[cur][bn * LPAD + bko];
#pragma unroll
        for (int k0 = 0; k0 < HID; k0 += 32) {
            AFrag fa, fb;
            fa.u[0] = *(const uint4*)(arow + k0 + kb);
            fa.u[1] = *(const uint4*)(arow + k0 + 8 + kb);
            fb.u[0] = *(const uint4*)(bbase + k0);
            fb.u[1] = *(const uint4*)(bbase + k0 + 8);
            acc = __builtin_amdgcn_wmma_f32_16x16x32_bf16(
                false, fa.v, false, fb.v, (short)0, acc, false, false);
        }

        int n = nt * 16 + (lane & 15);
        float bias = sb[n];
#pragma unroll
        for (int vv = 0; vv < 8; ++vv) {
            float logit = acc[vv] + bias;
            float om = rmax[vv];
            float nm = fmaxf(om, logit);
            rsum[vv] = rsum[vv] * __expf(om - nm) + __expf(logit - nm);
            rmax[vv] = nm;
            if (n == mylab[vv]) rlab[vv] = logit;
        }

        __syncthreads();                                // all waves done with smem[cur]
        if (nt + 1 < NTILES) {
#pragma unroll
            for (int i = 0; i < 8; ++i) {
                int q = tid * 8 + i;
                *(uint4*)&smem[(nt + 1) & 1][(q >> 6) * LPAD + (q & 63) * 8] = stage[i];
            }
            __syncthreads();                            // next tile visible
        }
    }

    // Merge the 16 per-lane partials of each row (xor masks 1..8 stay within
    // each 16-lane half; rows m<8 live in lanes 0-15, m>=8 in lanes 16-31).
#pragma unroll
    for (int vv = 0; vv < 8; ++vv) {
        float m = rmax[vv], s = rsum[vv], lb = rlab[vv];
        for (int off = 1; off < 16; off <<= 1) {
            float m2 = __shfl_xor(m, off, 32);
            float s2 = __shfl_xor(s, off, 32);
            float l2 = __shfl_xor(lb, off, 32);
            float nm = fmaxf(m, m2);
            s = s * __expf(m - nm) + s2 * __expf(m2 - nm);
            m = nm;
            lb = fmaxf(lb, l2);
        }
        if ((lane & 15) == 0) {
            int r = r0 + vv + mofs;
            xent[r] = (m + __logf(s)) - lb;             // logsumexp - label logit
        }
    }
}

// ---------------------------------------------------------------------------
// Final masked sequence loss: per-t weighted mean over batch, then mean over t.
// ---------------------------------------------------------------------------
__global__ void loss_reduce(const float* __restrict__ xent,      // [T*B], r=t*B+b
                            const float* __restrict__ seq_mask,  // [B, T]
                            float* __restrict__ out)
{
    __shared__ float sh[SEQT];
    int t = threadIdx.x;
    if (t < SEQT) {
        float num = 0.0f, den = 0.0f;
        for (int b = 0; b < BATCH; ++b) {
            float w = seq_mask[b * SEQT + t];
            num += xent[t * BATCH + b] * w;
            den += w;
        }
        sh[t] = num / (den + 1e-12f);
    }
    __syncthreads();
    if (t == 0) {
        float s = 0.0f;
        for (int i = 0; i < SEQT; ++i) s += sh[i];
        *out = s / (float)SEQT;
    }
}

// ---------------------------------------------------------------------------
// Host orchestration
// ---------------------------------------------------------------------------
extern "C" void kernel_launch(void* const* d_in, const int* in_sizes, int n_in,
                              void* d_out, int out_size, void* d_ws, size_t ws_size,
                              hipStream_t stream)
{
    const int*   features  = (const int*)  d_in[0];
    const int*   labels    = (const int*)  d_in[1];
    const int*   seqlen    = (const int*)  d_in[2];
    const float* seq_mask  = (const float*)d_in[3];
    const float* embedding = (const float*)d_in[4];
    const float* W0x = (const float*)d_in[5];
    const float* W0h = (const float*)d_in[6];
    const float* b0  = (const float*)d_in[7];
    const float* W1x = (const float*)d_in[8];
    const float* W1h = (const float*)d_in[9];
    const float* b1  = (const float*)d_in[10];
    const float* SW  = (const float*)d_in[11];
    const float* Sb  = (const float*)d_in[12];

    char* p = (char*)d_ws;
    auto carve = [&](size_t bytes) -> char* {
        char* r = p; p += (bytes + 255) & ~(size_t)255; return r;
    };
    uint16_t* W0T  = (uint16_t*)carve((size_t)G4 * KCAT * 2);
    uint16_t* W1T  = (uint16_t*)carve((size_t)G4 * KCAT * 2);
    uint16_t* SWT  = (uint16_t*)carve((size_t)VOCAB * HID * 2);
    uint16_t* xbf  = (uint16_t*)carve((size_t)SEQT * BATCH * HID * 2);
    uint16_t* outs = (uint16_t*)carve((size_t)SEQT * BATCH * HID * 2);
    float*    states = (float*)carve((size_t)4 * BATCH * HID * 4);   // h0,c0,h1,c1
    uint16_t* hbf  = (uint16_t*)carve((size_t)4 * BATCH * HID * 2);  // h0 A/B, h1 A/B
    uint16_t* o0bf = (uint16_t*)carve((size_t)BATCH * HID * 2);
    float*    xent = (float*)carve((size_t)SEQT * BATCH * 4);

    float* h0 = states;
    float* c0 = states + (size_t)BATCH * HID;
    float* h1 = states + (size_t)2 * BATCH * HID;
    float* c1 = states + (size_t)3 * BATCH * HID;
    uint16_t* h0A = hbf;
    uint16_t* h0B = hbf + (size_t)BATCH * HID;
    uint16_t* h1A = hbf + (size_t)2 * BATCH * HID;
    uint16_t* h1B = hbf + (size_t)3 * BATCH * HID;

    // Zero recurrent state (fp32 h/c + all bf16 recurrent buffers; contiguous)
    {
        int ndw = (4 * BATCH * HID * 4 + 4 * BATCH * HID * 2) / 4;
        zero_u32<<<(ndw + 255) / 256, 256, 0, stream>>>((uint32_t*)states, ndw);
    }
    pack_lstm_weights<<<(G4 * KCAT) / 256, 256, 0, stream>>>(W0x, W0h, W0T);
    pack_lstm_weights<<<(G4 * KCAT) / 256, 256, 0, stream>>>(W1x, W1h, W1T);
    pack_softmax_w<<<(VOCAB * HID) / 256, 256, 0, stream>>>(SW, SWT);
    embed_gather<<<(SEQT * BATCH * HID) / 256, 256, 0, stream>>>(features, embedding, xbf);

    uint16_t* h0in = h0A; uint16_t* h0out = h0B;
    uint16_t* h1in = h1A; uint16_t* h1out = h1B;
    for (int t = 0; t < SEQT; ++t) {
        lstm_cell<<<HID / 16, 256, 0, stream>>>(xbf + (size_t)t * BATCH * HID,
                                                h0in, W0T, b0, h0, c0, h0out, o0bf,
                                                seqlen, t);
        lstm_cell<<<HID / 16, 256, 0, stream>>>(o0bf,
                                                h1in, W1T, b1, h1, c1, h1out,
                                                outs + (size_t)t * BATCH * HID,
                                                seqlen, t);
        uint16_t* tmp;
        tmp = h0in; h0in = h0out; h0out = tmp;
        tmp = h1in; h1in = h1out; h1out = tmp;
    }

    logits_xent<<<(SEQT * BATCH / 16) / 4, 128, 0, stream>>>(outs, SWT, Sb, labels, xent);
    loss_reduce<<<1, 64, 0, stream>>>(xent, seq_mask, (float*)d_out);
}
